// MambaLayer_40209483825880
// MI455X (gfx1250) — compile-verified, matrix operand order checked by reading, not run on hardware
//
#include <hip/hip_runtime.h>
#include <hip/hip_bf16.h>
#include <math.h>

// ---------------------------------------------------------------------------
// Multimodal Mamba fusion forward for MI455X (gfx1250, wave32, WMMA).
//
// d_in layout assumption (jax tree-flatten, dicts sorted by key):
//   0:text 1:text_mask 2:audio 3:audio_mask 4:image 5:image_mask
//   then params leaves:
//   ad_a@6 ad_i@12 ad_t@18        (db,dw,ln_b,ln_g,ub,uw)
//   ca_a@24 ca_i@34               (kb,kw,ln_b,ln_g,ob,ow,qb,qw,vb,vw)
//   g_a@44 g_i@48                 (b1,b2,w1,w2)
//   g_ln_a_b@52 g_ln_a_g@53 g_ln_i_b@54 g_ln_i_g@55
//   mb_a@56 mb_f@67 mb_i@78 mb_t@89 (A_log,D,conv_b,conv_w,dt_b,dt_w,in_w,
//                                    ln_b,ln_g,out_w,x_w)
//   proj_b@100 proj_ln_b@101 proj_ln_g@102 proj_w@103
//
// Strategy: all dense projections run through v_wmma_f32_16x16x32_f16 with
// f32 accumulation.  A and W are pre-repacked to f16 scratch once per GEMM
// (hoists all cvt out of the hot loop; halves operand bytes).  Selective scan
// is one wave per (b,channel), 64 states = 2/lane, shfl_xor reduction.
// Cross-attention is fused per (b,head) in LDS with two WMMA stages.
// ---------------------------------------------------------------------------

typedef __attribute__((ext_vector_type(16))) _Float16 v16h;
typedef __attribute__((ext_vector_type(8)))  _Float16 v8h;
typedef __attribute__((ext_vector_type(8)))  float    v8f;

#define ACT_NONE 0
#define ACT_RELU 1
#define ACT_SILU 2
#define ACT_SOFTPLUS 3

__device__ __forceinline__ float act_apply(float v, int act) {
  if (act == ACT_RELU)     return v > 0.f ? v : 0.f;
  if (act == ACT_SILU)     return v / (1.f + expf(-v));
  if (act == ACT_SOFTPLUS) return (v > 20.f) ? v : log1pf(expf(v));
  return v;
}

__device__ __forceinline__ float wave_sum(float v) {
#pragma unroll
  for (int m = 16; m > 0; m >>= 1) v += __shfl_xor(v, m, 32);
  return v;
}
__device__ __forceinline__ float wave_max(float v) {
#pragma unroll
  for (int m = 16; m > 0; m >>= 1) v = fmaxf(v, __shfl_xor(v, m, 32));
  return v;
}

// WMMA f16 operand K-pattern (V_WMMA_F32_16X16X32_F16):
// lanes0-15: K = {k0..k0+7, k0+16..k0+23}; lanes16-31: the +8 variants.
// koff = (lane&16)?8:0.  All K's here are multiples of 16, so:
//  - low 8-half chunk (k0+koff)    is valid whenever k0 < K
//  - high 8-half chunk (k0+16+koff) is valid iff k0+32 <= K
// => guard-free main loop over K&~31, one half-step if (K&16).

__device__ __forceinline__ v16h merge16(v8h lo, v8h hi) {
  v16h r;
#pragma unroll
  for (int j = 0; j < 8; ++j) { r[j] = lo[j]; r[8 + j] = hi[j]; }
  return r;
}

// f16 operand fragment: two aligned 16B loads (works for global or LDS ptrs).
__device__ __forceinline__ v16h hload16(const _Float16* p, int k0, int koff) {
  return merge16(*(const v8h*)(p + k0 + koff),
                 *(const v8h*)(p + k0 + 16 + koff));
}
__device__ __forceinline__ v16h hload16_half(const _Float16* p, int k0, int koff) {
  v8h lo = *(const v8h*)(p + k0 + koff);
  v16h r;
#pragma unroll
  for (int j = 0; j < 8; ++j) { r[j] = lo[j]; r[8 + j] = (_Float16)0.f; }
  return r;
}

// Strided f32 -> dense f16 repack: dst[r*cols+c] = src[r*lds+c].
__global__ void cvt_f16_kernel(const float* __restrict__ src, int lds,
                               _Float16* __restrict__ dst, int cols, int n) {
  int i = blockIdx.x * blockDim.x + threadIdx.x;
  if (i >= n) return;
  int r = i / cols, c = i % cols;
  dst[i] = (_Float16)src[(size_t)r * lds + c];
}

// ---------------------------------------------------------------------------
// C[M,N] = act(Af16[M,K] * Wf16[N,K]^T + bias) + resid.
// One wave per 16x64 strip (4 N-subtiles share one A fragment).
// Subtile enables are blockIdx-derived (scalar-uniform): EXEC stays all-ones
// through every WMMA; disabled subtiles just skip the store.
// ---------------------------------------------------------------------------
__global__ void gemm_wmma_kernel(const _Float16* __restrict__ A,
                                 const _Float16* __restrict__ W,
                                 const float* __restrict__ bias,
                                 const float* __restrict__ resid, int ldr,
                                 float* __restrict__ C, int ldc,
                                 int M, int N, int K, int act) {
  int lane = threadIdx.x & 31;
  int tm  = blockIdx.y * 16;
  int tn0 = blockIdx.x * 64;
  int koff = (lane & 16) ? 8 : 0;
  const _Float16* Arow = A + (size_t)(tm + (lane & 15)) * K;

  const _Float16* Wr[4];
  bool en[4];
#pragma unroll
  for (int j = 0; j < 4; ++j) {
    int tn = tn0 + 16 * j;
    en[j] = (tn < N);                       // uniform across the wave
    Wr[j] = W + (size_t)((en[j] ? tn : 0) + (lane & 15)) * K;
  }

  v8f z = {};
  v8f acc[4];
#pragma unroll
  for (int j = 0; j < 4; ++j) acc[j] = z;

  int K32 = K & ~31;
  for (int k0 = 0; k0 < K32; k0 += 32) {
    __builtin_prefetch(Arow + k0 + 256, 0, 0);   // global_prefetch path
    v16h a = hload16(Arow, k0, koff);
#pragma unroll
    for (int j = 0; j < 4; ++j) {
      v16h b = hload16(Wr[j], k0, koff);
      acc[j] = __builtin_amdgcn_wmma_f32_16x16x32_f16(false, a, false, b,
                                                      (short)0, acc[j],
                                                      false, false);
    }
  }
  if (K & 16) {                                  // only the K=48 dt-projection
    v16h a = hload16_half(Arow, K32, koff);
#pragma unroll
    for (int j = 0; j < 4; ++j) {
      v16h b = hload16_half(Wr[j], K32, koff);
      acc[j] = __builtin_amdgcn_wmma_f32_16x16x32_f16(false, a, false, b,
                                                      (short)0, acc[j],
                                                      false, false);
    }
  }

  int rbase = tm + ((lane & 16) ? 8 : 0);
#pragma unroll
  for (int j = 0; j < 4; ++j) {
    if (!en[j]) continue;
    int col = tn0 + 16 * j + (lane & 15);
    float bv = bias ? bias[col] : 0.f;
#pragma unroll
    for (int r = 0; r < 8; ++r) {
      int row = rbase + r;
      float v = act_apply(acc[j][r] + bv, act);
      if (resid) v += resid[(size_t)row * ldr + col];
      C[(size_t)row * ldc + col] = v;
    }
  }
}

// ---------------------------------------------------------------------------
// LayerNorm over D per row (one wave/row); optional add before normalize.
// ---------------------------------------------------------------------------
__global__ void ln_kernel(const float* __restrict__ x, const float* __restrict__ add,
                          const float* __restrict__ g, const float* __restrict__ b,
                          float* __restrict__ y, int D, float eps) {
  size_t row = blockIdx.x;
  int lane = threadIdx.x;
  const float* xr = x + row * D;
  const float* ar = add ? add + row * D : nullptr;
  float s = 0.f;
  for (int i = lane; i < D; i += 32) s += xr[i] + (ar ? ar[i] : 0.f);
  float mu = wave_sum(s) / (float)D;
  float v2 = 0.f;
  for (int i = lane; i < D; i += 32) {
    float v = xr[i] + (ar ? ar[i] : 0.f) - mu;
    v2 += v * v;
  }
  float inv = rsqrtf(wave_sum(v2) / (float)D + eps);
  for (int i = lane; i < D; i += 32) {
    float v = xr[i] + (ar ? ar[i] : 0.f);
    y[row * D + i] = g[i] * (v - mu) * inv + b[i];
  }
}

// Causal depthwise conv1d (k=4) + bias + SiLU.  x is cols [0,1536) of xz.
__global__ void conv_silu_kernel(const float* __restrict__ xz, int ld,
                                 const float* __restrict__ cw,
                                 const float* __restrict__ cb,
                                 float* __restrict__ xc, int B, int L) {
  int idx = blockIdx.x * blockDim.x + threadIdx.x;
  int total = B * L * 1536;
  if (idx >= total) return;
  int d = idx % 1536;
  int l = (idx / 1536) % L;
  int b = idx / (1536 * L);
  float s = cb[d];
#pragma unroll
  for (int k = 0; k < 4; ++k) {
    int ls = l - 3 + k;
    if (ls >= 0) s += xz[(size_t)(b * L + ls) * ld + d] * cw[d * 4 + k];
  }
  xc[idx] = s / (1.f + expf(-s));
}

// Selective scan: one wave per (b, channel d); 64 states = 2 per lane.
__global__ void scan_kernel(const float* __restrict__ xc,
                            const float* __restrict__ dt,
                            const float* __restrict__ xdbl,
                            const float* __restrict__ Alog,
                            const float* __restrict__ Dv,
                            float* __restrict__ y, int B, int L) {
  const int di = 1536, ds = 64;
  int wid = (blockIdx.x * blockDim.x + threadIdx.x) >> 5;
  int lane = threadIdx.x & 31;
  if (wid >= B * di) return;
  int b = wid / di, d = wid % di;
  int s0 = lane, s1 = lane + 32;
  float A0 = -expf(Alog[d * ds + s0]);
  float A1 = -expf(Alog[d * ds + s1]);
  float h0 = 0.f, h1 = 0.f;
  float Dd = Dv[d];
  for (int t = 0; t < L; ++t) {
    size_t base = (size_t)(b * L + t);
    float dtv = dt[base * di + d];
    float xv  = xc[base * di + d];
    const float* xr = xdbl + base * 176;
    float B0 = xr[48 + s0], B1 = xr[48 + s1];
    float C0 = xr[112 + s0], C1 = xr[112 + s1];
    float du = dtv * xv;
    h0 = h0 * expf(dtv * A0) + du * B0;
    h1 = h1 * expf(dtv * A1) + du * B1;
    float part = wave_sum(h0 * C0 + h1 * C1);
    if (lane == 0) y[base * di + d] = part + xv * Dd;
  }
}

// y *= silu(z)   (z = cols [1536,3072) of xz)
__global__ void silu_gate_kernel(float* __restrict__ y,
                                 const float* __restrict__ xz, int M) {
  int idx = blockIdx.x * blockDim.x + threadIdx.x;
  if (idx >= M * 1536) return;
  int row = idx / 1536, d = idx % 1536;
  float z = xz[(size_t)row * 3072 + 1536 + d];
  y[idx] *= z / (1.f + expf(-z));
}

// ---------------------------------------------------------------------------
// Fused cross-attention head: one block per (b,h).  Lq=64, Lk=96, hd=64.
// scores = (Q Kt)/8 + maskbias -> softmax -> ctx = P V.  Both GEMMs via WMMA.
// ---------------------------------------------------------------------------
__global__ void attn_kernel(const float* __restrict__ Q,
                            const float* __restrict__ Kv,
                            const float* __restrict__ V,
                            const float* __restrict__ mask,
                            float* __restrict__ ctxout, int ldq) {
  const int LQ = 64, LK = 96, HD = 64;
  int b = blockIdx.y, h = blockIdx.x;
  int tid = threadIdx.x, w = tid >> 5, lane = tid & 31;

  __shared__ __align__(16) unsigned char smem[61440];
  _Float16* qs = (_Float16*)smem;                     // 64x64 f16 (8KB)
  _Float16* ks = (_Float16*)(smem + 8192);            // 96x64 f16 (12KB)
  float*    sc = (float*)(smem + 24576);              // 64x96 f32 (24KB)
  _Float16* vt = (_Float16*)(smem + 49152);           // 64x96 f16 (12KB) V^T
  _Float16* ps = (_Float16*)smem;                     // 64x96 f16, overlays qs/ks

  for (int i = tid; i < LQ * HD; i += 256) {
    int r = i >> 6, c = i & 63;
    qs[i] = (_Float16)Q[(size_t)(b * LQ + r) * ldq + h * HD + c];
  }
  for (int i = tid; i < LK * HD; i += 256) {
    int r = i >> 6, c = i & 63;
    float vv = V[(size_t)(b * LK + r) * ldq + h * HD + c];
    ks[i] = (_Float16)Kv[(size_t)(b * LK + r) * ldq + h * HD + c];
    vt[c * LK + r] = (_Float16)vv;
  }
  __syncthreads();

  int koff = (lane & 16) ? 8 : 0;
  // scores: 4x6 tiles of 16x16, 3 tiles per wave
  for (int t = w; t < 24; t += 8) {
    int mi = t / 6, ni = t % 6;
    v8f acc = {};
    for (int k0 = 0; k0 < HD; k0 += 32) {
      v16h a  = hload16(qs + (mi * 16 + (lane & 15)) * HD, k0, koff);
      v16h bb = hload16(ks + (ni * 16 + (lane & 15)) * HD, k0, koff);
      acc = __builtin_amdgcn_wmma_f32_16x16x32_f16(false, a, false, bb,
                                                   (short)0, acc, false, false);
    }
#pragma unroll
    for (int r = 0; r < 8; ++r) {
      int row = mi * 16 + r + ((lane & 16) ? 8 : 0);
      int col = ni * 16 + (lane & 15);
      float mb = (mask[b * LK + col] == 0.f) ? -1e30f : 0.f;
      sc[row * LK + col] = acc[r] * 0.125f + mb;
    }
  }
  __syncthreads();

  // softmax: 8 rows per wave; write probs as f16 into ps (qs/ks now dead)
  for (int row = w; row < LQ; row += 8) {
    float m = -1e30f;
    for (int c = lane; c < LK; c += 32) m = fmaxf(m, sc[row * LK + c]);
    m = wave_max(m);
    float s = 0.f;
    for (int c = lane; c < LK; c += 32) {
      float e = expf(sc[row * LK + c] - m);
      sc[row * LK + c] = e;
      s += e;
    }
    float inv = 1.f / wave_sum(s);
    for (int c = lane; c < LK; c += 32)
      ps[row * LK + c] = (_Float16)(sc[row * LK + c] * inv);
  }
  __syncthreads();

  // ctx = P(64x96) @ V(96x64): 4x4 tiles, 2 per wave; B-operand from V^T rows
  for (int t = w; t < 16; t += 8) {
    int mi = t >> 2, ni = t & 3;
    v8f acc = {};
    for (int k0 = 0; k0 < LK; k0 += 32) {
      v16h a  = hload16(ps + (mi * 16 + (lane & 15)) * LK, k0, koff);
      v16h bb = hload16(vt + (ni * 16 + (lane & 15)) * LK, k0, koff);
      acc = __builtin_amdgcn_wmma_f32_16x16x32_f16(false, a, false, bb,
                                                   (short)0, acc, false, false);
    }
#pragma unroll
    for (int r = 0; r < 8; ++r) {
      int row = mi * 16 + r + ((lane & 16) ? 8 : 0);
      int col = ni * 16 + (lane & 15);
      ctxout[(size_t)(b * LQ + row) * ldq + h * HD + col] = acc[r];
    }
  }
}

// gate second layer: g[row] = sigmoid(dot(h1[row,:256], w2) + b2). Wave/row.
__global__ void gate2_kernel(const float* __restrict__ h1,
                             const float* __restrict__ w2,
                             const float* __restrict__ b2,
                             float* __restrict__ g, int M) {
  int row = blockIdx.x, lane = threadIdx.x;
  float s = 0.f;
  for (int j = lane; j < 256; j += 32) s += h1[(size_t)row * 256 + j] * w2[j];
  s = wave_sum(s);
  if (lane == 0) g[row] = 1.f / (1.f + expf(-(s + b2[0])));
}

// out = g*a + (1-g)*b   (g broadcast per row of 768)
__global__ void fuse_kernel(const float* __restrict__ g,
                            const float* __restrict__ a,
                            const float* __restrict__ bb,
                            float* __restrict__ out, int M) {
  int idx = blockIdx.x * blockDim.x + threadIdx.x;
  if (idx >= M * 768) return;
  float gg = g[idx / 768];
  out[idx] = gg * a[idx] + (1.f - gg) * bb[idx];
}

__global__ void concat_kernel(const float* __restrict__ a,
                              const float* __restrict__ b,
                              const float* __restrict__ c,
                              float* __restrict__ out, int M) {
  int idx = blockIdx.x * blockDim.x + threadIdx.x;
  if (idx >= M * 2304) return;
  int row = idx / 2304, col = idx % 2304;
  float v;
  if (col < 768)        v = a[(size_t)row * 768 + col];
  else if (col < 1536)  v = b[(size_t)row * 768 + col - 768];
  else                  v = c[(size_t)row * 768 + col - 1536];
  out[idx] = v;
}

__global__ void mean_kernel(const float* __restrict__ x, float* __restrict__ out,
                            int B, int L, int D) {
  int idx = blockIdx.x * blockDim.x + threadIdx.x;
  if (idx >= B * D) return;
  int b = idx / D, c = idx % D;
  float s = 0.f;
  for (int l = 0; l < L; ++l) s += x[(size_t)(b * L + l) * D + c];
  out[idx] = s / (float)L;
}

// ---------------------------------------------------------------------------
// Host-side orchestration
// ---------------------------------------------------------------------------
static inline int cdiv(int a, int b) { return (a + b - 1) / b; }

struct Scratch {
  float *lnb, *xz, *xc, *xdbl, *dt, *ys;   // mamba f32 scratch
  _Float16 *aS, *wS;                       // f16 operand repack scratch
};

// Repack A (strided f32) and W (dense f32) to f16, then WMMA GEMM.
static void gemm(hipStream_t st, const float* A, int lda, const float* W,
                 const float* bias, const float* resid, int ldr,
                 float* C, int ldc, int M, int N, int K, int act,
                 const Scratch& sc) {
  cvt_f16_kernel<<<cdiv(M * K, 256), 256, 0, st>>>(A, lda, sc.aS, K, M * K);
  cvt_f16_kernel<<<cdiv(N * K, 256), 256, 0, st>>>(W, K, sc.wS, K, N * K);
  gemm_wmma_kernel<<<dim3(cdiv(N, 64), M / 16), 32, 0, st>>>(
      sc.aS, sc.wS, bias, resid, ldr, C, ldc, M, N, K, act);
}
static void ln(hipStream_t st, const float* x, const float* add, const float* g,
               const float* b, float* y, int rows, int D, float eps) {
  ln_kernel<<<rows, 32, 0, st>>>(x, add, g, b, y, D, eps);
}

static void run_mamba(hipStream_t st, const float* in, float* out, int L,
                      void* const* d_in, int base, const Scratch& s) {
  const float* A_log = (const float*)d_in[base + 0];
  const float* Dv    = (const float*)d_in[base + 1];
  const float* cb    = (const float*)d_in[base + 2];
  const float* cw    = (const float*)d_in[base + 3];
  const float* dt_b  = (const float*)d_in[base + 4];
  const float* dt_w  = (const float*)d_in[base + 5];
  const float* in_w  = (const float*)d_in[base + 6];
  const float* ln_b  = (const float*)d_in[base + 7];
  const float* ln_g  = (const float*)d_in[base + 8];
  const float* out_w = (const float*)d_in[base + 9];
  const float* x_w   = (const float*)d_in[base + 10];
  const int B = 4, M = B * L;
  ln(st, in, nullptr, ln_g, ln_b, s.lnb, M, 768, 1e-5f);
  gemm(st, s.lnb, 768, in_w, nullptr, nullptr, 0, s.xz, 3072, M, 3072, 768, ACT_NONE, s);
  conv_silu_kernel<<<cdiv(M * 1536, 256), 256, 0, st>>>(s.xz, 3072, cw, cb, s.xc, B, L);
  gemm(st, s.xc, 1536, x_w, nullptr, nullptr, 0, s.xdbl, 176, M, 176, 1536, ACT_NONE, s);
  gemm(st, s.xdbl, 176, dt_w, dt_b, nullptr, 0, s.dt, 1536, M, 1536, 48, ACT_SOFTPLUS, s);
  scan_kernel<<<cdiv(B * 1536, 8), 256, 0, st>>>(s.xc, s.dt, s.xdbl, A_log, Dv, s.ys, B, L);
  silu_gate_kernel<<<cdiv(M * 1536, 256), 256, 0, st>>>(s.ys, s.xz, M);
  gemm(st, s.ys, 1536, out_w, nullptr, in, 768, out, 768, M, 768, 1536, ACT_NONE, s);
}

extern "C" void kernel_launch(void* const* d_in, const int* in_sizes, int n_in,
                              void* d_out, int out_size, void* d_ws, size_t ws_size,
                              hipStream_t stream) {
  (void)in_sizes; (void)n_in; (void)out_size; (void)ws_size;
  hipStream_t st = stream;
#define PF(i) ((const float*)d_in[i])
  const float* text  = PF(0);
  const float* audio = PF(2);
  const float* amask = PF(3);
  const float* image = PF(4);
  const float* imask = PF(5);
  enum { AD_A = 6, AD_I = 12, AD_T = 18, CA_A = 24, CA_I = 34, G_A = 44, G_I = 48,
         G_LN_A_B = 52, G_LN_A_G = 53, G_LN_I_B = 54, G_LN_I_G = 55,
         MB_A = 56, MB_F = 67, MB_I = 78, MB_T = 89,
         PROJ_B = 100, PROJ_LN_B = 101, PROJ_LN_G = 102, PROJ_W = 103 };

  // workspace bump allocator (256B aligned)
  char* ws = (char*)d_ws;
  size_t off = 0;
  auto allocb = [&](size_t nbytes) -> void* {
    void* p = (void*)(ws + off);
    off += (nbytes + 255) & ~(size_t)255;
    return p;
  };
  auto alloc = [&](size_t nfloats) -> float* {
    return (float*)allocb(nfloats * sizeof(float));
  };

  const int Mt = 256, Ma = 384;  // 4*64, 4*96
  Scratch ms;
  ms.lnb  = alloc(Ma * 768);
  ms.xz   = alloc((size_t)Ma * 3072);
  ms.xc   = alloc((size_t)Ma * 1536);
  ms.xdbl = alloc(Ma * 176);
  ms.dt   = alloc((size_t)Ma * 1536);
  ms.ys   = alloc((size_t)Ma * 1536);
  ms.aS   = (_Float16*)allocb((size_t)589824 * 2);    // max M*K halves
  ms.wS   = (_Float16*)allocb((size_t)2359296 * 2);   // max N*K halves
  float* h96  = alloc(Ma * 96);    // adapter bottleneck
  float* ad_t = alloc(Mt * 768);
  float* ad_a = alloc(Ma * 768);
  float* ad_i = alloc(Ma * 768);
  float* tm   = alloc(Mt * 768);
  float* am   = alloc(Ma * 768);
  float* im   = alloc(Ma * 768);
  float* qb   = alloc(Mt * 768);
  float* kb   = alloc(Ma * 768);
  float* vb   = alloc(Ma * 768);
  float* ctx  = alloc(Mt * 768);
  float* ob   = alloc(Mt * 768);
  float* ta   = alloc(Mt * 768);
  float* ti   = alloc(Mt * 768);
  float* g1   = alloc(Mt * 256);
  float* ga   = alloc(Mt);
  float* gi   = alloc(Mt);
  float* ftmp = alloc(Mt * 768);
  float* ares = alloc(Mt * 768);
  float* ires = alloc(Mt * 768);
  float* cat  = alloc((size_t)Mt * 2304);
  float* fp   = alloc(Mt * 768);
  float* fln  = alloc(Mt * 768);
  float* fmb  = alloc(Mt * 768);

  // --- adapters: out = x + relu(LN(x)@dw^T+db)@uw^T + ub
  auto adapter = [&](const float* in, float* out, int M, int base) {
    ln(st, in, nullptr, PF(base + 3), PF(base + 2), ms.lnb, M, 768, 1e-5f);
    gemm(st, ms.lnb, 768, PF(base + 1), PF(base + 0), nullptr, 0,
         h96, 96, M, 96, 768, ACT_RELU, ms);
    gemm(st, h96, 96, PF(base + 5), PF(base + 4), in, 768,
         out, 768, M, 768, 96, ACT_NONE, ms);
  };
  adapter(text,  ad_t, Mt, AD_T);
  adapter(audio, ad_a, Ma, AD_A);
  adapter(image, ad_i, Ma, AD_I);

  // --- mamba blocks
  run_mamba(st, ad_t, tm, 64, d_in, MB_T, ms);
  run_mamba(st, ad_a, am, 96, d_in, MB_A, ms);
  run_mamba(st, ad_i, im, 96, d_in, MB_I, ms);

  // --- cross attention (audio / image), reusing q/k/v/ctx/ob scratch
  auto cross = [&](const float* kv, const float* mask, int base, float* out) {
    gemm(st, tm, 768, PF(base + 7), PF(base + 6), nullptr, 0, qb, 768,
         Mt, 768, 768, ACT_NONE, ms);
    gemm(st, kv, 768, PF(base + 1), PF(base + 0), nullptr, 0, kb, 768,
         Ma, 768, 768, ACT_NONE, ms);
    gemm(st, kv, 768, PF(base + 9), PF(base + 8), nullptr, 0, vb, 768,
         Ma, 768, 768, ACT_NONE, ms);
    attn_kernel<<<dim3(12, 4), 256, 0, st>>>(qb, kb, vb, mask, ctx, 768);
    gemm(st, ctx, 768, PF(base + 5), PF(base + 4), tm, 768, ob, 768,
         Mt, 768, 768, ACT_NONE, ms);
    ln(st, ob, nullptr, PF(base + 3), PF(base + 2), out, Mt, 768, 1e-12f);
  };
  cross(am, amask, CA_A, ta);
  cross(im, imask, CA_I, ti);

  // --- gate MLPs
  auto gate = [&](int base, float* g) {
    gemm(st, tm, 768, PF(base + 2), PF(base + 0), nullptr, 0, g1, 256,
         Mt, 256, 768, ACT_RELU, ms);
    gate2_kernel<<<Mt, 32, 0, st>>>(g1, PF(base + 3), PF(base + 1), g, Mt);
  };
  gate(G_A, ga);
  gate(G_I, gi);

  // --- gated fusion + LN
  fuse_kernel<<<cdiv(Mt * 768, 256), 256, 0, st>>>(ga, ta, tm, ftmp, Mt);
  ln(st, ftmp, nullptr, PF(G_LN_A_G), PF(G_LN_A_B), ares, Mt, 768, 1e-5f);
  fuse_kernel<<<cdiv(Mt * 768, 256), 256, 0, st>>>(gi, ti, tm, ftmp, Mt);
  ln(st, ftmp, nullptr, PF(G_LN_I_G), PF(G_LN_I_B), ires, Mt, 768, 1e-5f);

  // --- concat -> proj -> LN -> fused mamba -> mean
  concat_kernel<<<cdiv(Mt * 2304, 256), 256, 0, st>>>(tm, ares, ires, cat, Mt);
  gemm(st, cat, 2304, PF(PROJ_W), PF(PROJ_B), nullptr, 0, fp, 768,
       Mt, 768, 2304, ACT_NONE, ms);
  ln(st, fp, nullptr, PF(PROJ_LN_G), PF(PROJ_LN_B), fln, Mt, 768, 1e-5f);
  run_mamba(st, fln, fmb, 64, d_in, MB_F, ms);
  mean_kernel<<<cdiv(4 * 768, 256), 256, 0, st>>>(fmb, (float*)d_out, 4, 64, 768);
#undef PF
}
